// RWKV7TimeMix_65025804861762
// MI455X (gfx1250) — compile-verified, matrix operand order checked by reading, not run on hardware
//
#include <hip/hip_runtime.h>

// ---------------------------------------------------------------------------
// RWKV7 TimeMix forward for MI455X (gfx1250, wave32, WMMA).
//
// Shapes: B=4, T=2048, C=2048, H=32, HS=64. ROWS = B*T = 8192.
// Heavy GEMMs (r/k/v/out projections, 2048x2048) + LoRA GEMMs use
// v_wmma_f32_16x16x32_bf16 (bf16 in, f32 accumulate) with double-buffered
// LDS staging (one barrier per K-step, global loads overlapped with WMMA).
// The sequential WKV7 scan uses 64-thread head-blocks with register-resident
// 64x64 state.
// ---------------------------------------------------------------------------

#define Bc   4
#define Tt   2048
#define Cc   2048
#define Hh   32
#define HS   64
#define ROWS (Bc * Tt)
#define NC   ((size_t)ROWS * Cc)
#define DW_  128
#define DA_  128
#define DG_  224
#define DV_  64
#define EPS_GN 0.00064f

typedef __attribute__((ext_vector_type(16))) __bf16 v16bf;
typedef __attribute__((ext_vector_type(8)))  float  v8f;

// float -> bf16 (round to nearest even)
__device__ __forceinline__ unsigned short f2bf(float f) {
    unsigned int u = __builtin_bit_cast(unsigned int, f);
    unsigned int r = u + 0x7FFFu + ((u >> 16) & 1u);
    return (unsigned short)(r >> 16);
}

__device__ __forceinline__ float sigmoidf_(float x) {
    return 1.0f / (1.0f + expf(-x));
}

// Load a 16-element bf16 fragment (32 bytes) from a (generic/LDS) pointer.
__device__ __forceinline__ v16bf ldfrag(const unsigned short* p) {
    struct P { uint4 a, b; } t;
    t.a = *(const uint4*)(p);
    t.b = *(const uint4*)(p + 8);
    return __builtin_bit_cast(v16bf, t);
}

// ---------------------------------------------------------------------------
// GEMM: D[M,N] (f32) = A[M,K] (bf16, row major) * W[N,K] (bf16, row major)^T
// Workgroup 256 threads = 8 waves; tile 128(M) x 128(N); K step 32.
// Wave grid 2x4: each wave owns a 64x32 sub-tile = 4x2 WMMA 16x16 tiles.
// Software-pipelined: global loads for tile i+1 are in flight while the
// WMMAs for tile i run out of LDS buffer (i&1); one barrier per K-step.
// M must be a multiple of 128; N guarded; K a multiple of 32.
// ---------------------------------------------------------------------------
#define TILE_M 128
#define TILE_N 128
#define TILE_K 32

__global__ __launch_bounds__(256)
void gemm_bf16_nt_kernel(const unsigned short* __restrict__ A,
                         const unsigned short* __restrict__ W,
                         float* __restrict__ D,
                         int M, int N, int K)
{
    __shared__ alignas(16) unsigned short As[2][TILE_M * TILE_K];
    __shared__ alignas(16) unsigned short Ws[2][TILE_N * TILE_K];

    const int tid   = threadIdx.x;
    const int lane  = tid & 31;
    const int wid   = tid >> 5;      // 0..7
    const int waveM = wid >> 2;      // 0..1  (64 rows each)
    const int waveN = wid & 3;       // 0..3  (32 cols each)

    const int m0 = blockIdx.y * TILE_M;
    const int n0 = blockIdx.x * TILE_N;

    v8f acc[4][2] = {};

    const int ldRow = tid >> 1;          // 0..127
    const int ldSeg = (tid & 1) * 16;    // 0 or 16 (bf16 elements)
    const int wOk   = (n0 + ldRow) < N;  // guard for short N tiles

    const unsigned short* aSrc = A + (size_t)(m0 + ldRow) * K + ldSeg;
    const unsigned short* wSrc = W + (size_t)(n0 + ldRow) * K + ldSeg;

    uint4 ra0, ra1, rw0, rw1;

    auto gload = [&](int k0) {
        const unsigned short* pa = aSrc + k0;
        ra0 = *(const uint4*)(pa);
        ra1 = *(const uint4*)(pa + 8);
        rw0 = make_uint4(0u, 0u, 0u, 0u);
        rw1 = make_uint4(0u, 0u, 0u, 0u);
        if (wOk) {
            const unsigned short* pw = wSrc + k0;
            rw0 = *(const uint4*)(pw);
            rw1 = *(const uint4*)(pw + 8);
        }
    };
    auto sstore = [&](int buf) {
        *(uint4*)&As[buf][ldRow * TILE_K + ldSeg]     = ra0;
        *(uint4*)&As[buf][ldRow * TILE_K + ldSeg + 8] = ra1;
        *(uint4*)&Ws[buf][ldRow * TILE_K + ldSeg]     = rw0;
        *(uint4*)&Ws[buf][ldRow * TILE_K + ldSeg + 8] = rw1;
    };

    // prologue: stage tile 0
    gload(0);
    sstore(0);
    __syncthreads();

    const int khalf = (lane >> 4) * 16;  // lane-group K offset inside tile
    const int rsel  = lane & 15;
    const int ntile = K / TILE_K;

    for (int i = 0; i < ntile; ++i) {
        const int cur = i & 1;

        // kick off global loads for the next tile (latency hidden by WMMAs)
        if (i + 1 < ntile) {
            gload((i + 1) * TILE_K);
            if (i + 2 < ntile) __builtin_prefetch(aSrc + (i + 2) * TILE_K, 0, 0);
        }

        // compute on the current LDS buffer
        v16bf aF[4], bF[2];
        #pragma unroll
        for (int mt = 0; mt < 4; ++mt)
            aF[mt] = ldfrag(&As[cur][(waveM * 64 + mt * 16 + rsel) * TILE_K + khalf]);
        #pragma unroll
        for (int nt = 0; nt < 2; ++nt)
            bF[nt] = ldfrag(&Ws[cur][(waveN * 32 + nt * 16 + rsel) * TILE_K + khalf]);

        #pragma unroll
        for (int mt = 0; mt < 4; ++mt)
            #pragma unroll
            for (int nt = 0; nt < 2; ++nt)
                acc[mt][nt] = __builtin_amdgcn_wmma_f32_16x16x32_bf16(
                    false, aF[mt], false, bF[nt],
                    (short)0, acc[mt][nt], false, false);

        // stage the next tile into the other buffer; single barrier per step
        if (i + 1 < ntile) {
            sstore((i + 1) & 1);
            __syncthreads();
        }
    }

    // ---- epilogue: C/D layout: VGPR i, lane l -> M = i + (l>>4)*8, N = l&15
    const int mrow = (lane >> 4) * 8;
    const int ncol = lane & 15;
    #pragma unroll
    for (int mt = 0; mt < 4; ++mt) {
        #pragma unroll
        for (int nt = 0; nt < 2; ++nt) {
            const int n = n0 + waveN * 32 + nt * 16 + ncol;
            if (n < N) {
                #pragma unroll
                for (int i = 0; i < 8; ++i) {
                    const int m = m0 + waveM * 64 + mt * 16 + mrow + i;
                    D[(size_t)m * N + n] = acc[mt][nt][i];
                }
            }
        }
    }
}

// ---------------------------------------------------------------------------
// Token-shift mixing: xx = x[t-1] - x[t];  X? = bf16(x + xx * mix?)
// ---------------------------------------------------------------------------
__global__ void mix_kernel(const float* __restrict__ x,
                           const float* __restrict__ mr, const float* __restrict__ mw,
                           const float* __restrict__ mk, const float* __restrict__ mv,
                           const float* __restrict__ ma, const float* __restrict__ mg,
                           unsigned short* __restrict__ XR, unsigned short* __restrict__ XW,
                           unsigned short* __restrict__ XK, unsigned short* __restrict__ XV,
                           unsigned short* __restrict__ XA, unsigned short* __restrict__ XG)
{
    size_t idx = (size_t)blockIdx.x * blockDim.x + threadIdx.x;
    if (idx >= NC) return;
    const int c   = (int)(idx % Cc);
    const size_t row = idx / Cc;
    const int t   = (int)(row % Tt);
    const float xv = x[idx];
    const float xp = (t > 0) ? x[idx - Cc] : 0.0f;
    const float xx = xp - xv;
    XR[idx] = f2bf(xv + xx * mr[c]);
    XW[idx] = f2bf(xv + xx * mw[c]);
    XK[idx] = f2bf(xv + xx * mk[c]);
    XV[idx] = f2bf(xv + xx * mv[c]);
    XA[idx] = f2bf(xv + xx * ma[c]);
    XG[idx] = f2bf(xv + xx * mg[c]);
}

// f32 [R,Cw] -> bf16, optionally transposed to [Cw,R]
__global__ void cvt_bf16_kernel(const float* __restrict__ src,
                                unsigned short* __restrict__ dst,
                                int R, int Cw, int trans)
{
    size_t idx = (size_t)blockIdx.x * blockDim.x + threadIdx.x;
    size_t total = (size_t)R * Cw;
    if (idx >= total) return;
    const int cw = (int)(idx % Cw);
    const int r  = (int)(idx / Cw);
    const unsigned short v = f2bf(src[idx]);
    if (trans) dst[(size_t)cw * R + r] = v;
    else       dst[idx] = v;
}

// f32 -> (activation) -> bf16.  mode: 0=none, 1=tanh, 2=sigmoid
__global__ void act_bf16_kernel(const float* __restrict__ src,
                                unsigned short* __restrict__ dst,
                                size_t n, int mode)
{
    size_t idx = (size_t)blockIdx.x * blockDim.x + threadIdx.x;
    if (idx >= n) return;
    float v = src[idx];
    if (mode == 1)      v = tanhf(v);
    else if (mode == 2) v = sigmoidf_(v);
    dst[idx] = f2bf(v);
}

__device__ __forceinline__ float block64_reduce(float* red, int lane, float v) {
    red[lane] = v;
    __syncthreads();
    #pragma unroll
    for (int st = 32; st > 0; st >>= 1) {
        if (lane < st) red[lane] += red[lane + st];
        __syncthreads();
    }
    float r = red[0];
    __syncthreads();
    return r;
}

// ---------------------------------------------------------------------------
// Fused post-projection pointwise stage, one (row, head) per 64-thread block:
//   decay = exp(-exp(-softplus(-(w0+hw2)) - 0.5))   (written over W2buf)
//   a     = sigmoid(a0 + ha2)                        (written over A2buf)
//   v     = v + (v_first - v) * sigmoid(v0 + hv2)    (in-place in Vbuf)
//   kk    = normalize_head(k * k_k)                  (-> KK)
//   k     = k * (1 + (a-1) * k_a)                    (in-place in Kbuf)
// ---------------------------------------------------------------------------
__global__ __launch_bounds__(64)
void post_kernel(float* __restrict__ W2buf, float* __restrict__ A2buf,
                 const float* __restrict__ V2buf,
                 float* __restrict__ Kbuf, float* __restrict__ Vbuf,
                 const float* __restrict__ vfirst, float* __restrict__ KK,
                 const float* __restrict__ w0, const float* __restrict__ a0,
                 const float* __restrict__ v0, const float* __restrict__ k_k,
                 const float* __restrict__ k_a)
{
    __shared__ float red[64];
    const int h    = blockIdx.x;
    const int lane = threadIdx.x;
    const size_t row = blockIdx.y;
    const int c    = h * HS + lane;
    const size_t g = row * Cc + c;

    const float kv  = Kbuf[g];
    const float kkr = kv * k_k[c];
    const float ss  = block64_reduce(red, lane, kkr * kkr);
    const float inv = 1.0f / fmaxf(sqrtf(ss), 1e-12f);
    KK[g] = kkr * inv;

    const float av = sigmoidf_(a0[c] + A2buf[g]);
    A2buf[g] = av;

    float wv = w0[c] + W2buf[g];
    const float z  = -wv;
    const float sp = (z > 20.0f) ? z : log1pf(expf(z));   // softplus(-w)
    wv = -sp - 0.5f;
    W2buf[g] = expf(-expf(wv));

    const float vv = Vbuf[g];
    Vbuf[g] = vv + (vfirst[g] - vv) * sigmoidf_(v0[c] + V2buf[g]);

    Kbuf[g] = kv * (1.0f + (av - 1.0f) * k_a[c]);
}

// ---------------------------------------------------------------------------
// WKV7 recurrence. One (head, batch) per 64-thread block. Thread i owns state
// row i (64 f32 registers). Per-step vectors broadcast via LDS.
//   sa    = state @ (-kk)
//   state = state*w + v k^T + sa (kk*a)^T
//   y     = state @ r
// ---------------------------------------------------------------------------
__global__ __launch_bounds__(64)
void wkv_kernel(const float* __restrict__ R,  const float* __restrict__ Wd,
                const float* __restrict__ Kc, const float* __restrict__ Vc,
                const float* __restrict__ KK, const float* __restrict__ Ac,
                float* __restrict__ Y)
{
    const int h = blockIdx.x;
    const int b = blockIdx.y;
    const int i = threadIdx.x;

    float s[HS];
    #pragma unroll
    for (int j = 0; j < HS; ++j) s[j] = 0.0f;

    __shared__ float rs[HS], ws[HS], ks[HS], kks[HS], as2[HS];

    for (int t = 0; t < Tt; ++t) {
        const size_t base = ((size_t)(b * Tt + t)) * Cc + h * HS;
        rs[i]  = R[base + i];
        ws[i]  = Wd[base + i];
        ks[i]  = Kc[base + i];
        kks[i] = KK[base + i];
        as2[i] = Ac[base + i];
        const float vi = Vc[base + i];
        __syncthreads();

        float sa = 0.0f;
        #pragma unroll
        for (int j = 0; j < HS; ++j) sa -= s[j] * kks[j];

        float out = 0.0f;
        #pragma unroll
        for (int j = 0; j < HS; ++j) {
            const float sj = s[j] * ws[j] + vi * ks[j] + sa * (kks[j] * as2[j]);
            s[j] = sj;
            out += sj * rs[j];
        }
        Y[base + i] = out;
        __syncthreads();
    }
}

// ---------------------------------------------------------------------------
// Per-head GroupNorm + bonus + gate -> bf16 input for the Wo projection.
// ---------------------------------------------------------------------------
__global__ __launch_bounds__(64)
void gn_out_kernel(const float* __restrict__ Y,  const float* __restrict__ R,
                   const float* __restrict__ Kc, const float* __restrict__ Vc,
                   const float* __restrict__ G,  const float* __restrict__ r_k,
                   const float* __restrict__ ln_w, const float* __restrict__ ln_b,
                   unsigned short* __restrict__ OUTG)
{
    __shared__ float red[64];
    const int h    = blockIdx.x;
    const int lane = threadIdx.x;
    const size_t row = blockIdx.y;
    const int c    = h * HS + lane;
    const size_t g = row * Cc + c;

    const float yv   = Y[g];
    const float mean = block64_reduce(red, lane, yv) * (1.0f / HS);
    const float d    = yv - mean;
    const float var  = block64_reduce(red, lane, d * d) * (1.0f / HS);
    const float dot  = block64_reduce(red, lane, R[g] * Kc[g] * r_k[c]);

    const float yn    = d * rsqrtf(var + EPS_GN) * ln_w[c] + ln_b[c];
    const float bonus = dot * Vc[g];
    OUTG[g] = f2bf((yn + bonus) * G[g]);
}

// ---------------------------------------------------------------------------
// Host-side orchestration
// ---------------------------------------------------------------------------
static inline void launch_cvt(const float* src, unsigned short* dst,
                              int R, int Cw, int trans, hipStream_t stream) {
    size_t total = (size_t)R * Cw;
    cvt_bf16_kernel<<<(unsigned)((total + 255) / 256), 256, 0, stream>>>(src, dst, R, Cw, trans);
}

static inline void launch_gemm(const unsigned short* A, const unsigned short* W,
                               float* D, int M, int N, int K, hipStream_t stream) {
    dim3 grid((N + TILE_N - 1) / TILE_N, M / TILE_M);
    gemm_bf16_nt_kernel<<<grid, 256, 0, stream>>>(A, W, D, M, N, K);
}

extern "C" void kernel_launch(void* const* d_in, const int* in_sizes, int n_in,
                              void* d_out, int out_size, void* d_ws, size_t ws_size,
                              hipStream_t stream)
{
    (void)in_sizes; (void)n_in; (void)out_size; (void)ws_size;

    const float* x      = (const float*)d_in[0];
    const float* vfirst = (const float*)d_in[1];
    const float* x_r    = (const float*)d_in[2];
    const float* x_w    = (const float*)d_in[3];
    const float* x_k    = (const float*)d_in[4];
    const float* x_v    = (const float*)d_in[5];
    const float* x_a    = (const float*)d_in[6];
    const float* x_g    = (const float*)d_in[7];
    const float* w0     = (const float*)d_in[8];
    const float* w1     = (const float*)d_in[9];
    const float* w2     = (const float*)d_in[10];
    const float* a0     = (const float*)d_in[11];
    const float* a1     = (const float*)d_in[12];
    const float* a2     = (const float*)d_in[13];
    const float* v0     = (const float*)d_in[14];
    const float* v1     = (const float*)d_in[15];
    const float* v2     = (const float*)d_in[16];
    const float* g1     = (const float*)d_in[17];
    const float* g2     = (const float*)d_in[18];
    const float* k_k    = (const float*)d_in[19];
    const float* k_a    = (const float*)d_in[20];
    const float* r_k    = (const float*)d_in[21];
    const float* Wr     = (const float*)d_in[22];
    const float* Wk     = (const float*)d_in[23];
    const float* Wv     = (const float*)d_in[24];
    const float* Wo     = (const float*)d_in[25];
    const float* ln_w   = (const float*)d_in[26];
    const float* ln_b   = (const float*)d_in[27];
    float* out          = (float*)d_out;

    // Workspace carve-out (256B aligned slabs).
    char* ws = (char*)d_ws;
    size_t off = 0;
    auto alloc = [&](size_t bytes) -> void* {
        void* p = ws + off;
        off += (bytes + 255) & ~(size_t)255;
        return p;
    };

    unsigned short* XR = (unsigned short*)alloc(NC * 2);
    unsigned short* XW = (unsigned short*)alloc(NC * 2);
    unsigned short* XK = (unsigned short*)alloc(NC * 2);
    unsigned short* XV = (unsigned short*)alloc(NC * 2);
    unsigned short* XA = (unsigned short*)alloc(NC * 2);
    unsigned short* XG = (unsigned short*)alloc(NC * 2);

    unsigned short* WrB = (unsigned short*)alloc((size_t)Cc * Cc * 2);
    unsigned short* WkB = (unsigned short*)alloc((size_t)Cc * Cc * 2);
    unsigned short* WvB = (unsigned short*)alloc((size_t)Cc * Cc * 2);
    unsigned short* WoB = (unsigned short*)alloc((size_t)Cc * Cc * 2);

    unsigned short* w1B = (unsigned short*)alloc((size_t)Cc * DW_ * 2);
    unsigned short* w2B = (unsigned short*)alloc((size_t)Cc * DW_ * 2);
    unsigned short* a1B = (unsigned short*)alloc((size_t)Cc * DA_ * 2);
    unsigned short* a2B = (unsigned short*)alloc((size_t)Cc * DA_ * 2);
    unsigned short* v1B = (unsigned short*)alloc((size_t)Cc * DV_ * 2);
    unsigned short* v2B = (unsigned short*)alloc((size_t)Cc * DV_ * 2);
    unsigned short* g1B = (unsigned short*)alloc((size_t)Cc * DG_ * 2);
    unsigned short* g2B = (unsigned short*)alloc((size_t)Cc * DG_ * 2);

    float* Rb  = (float*)alloc(NC * 4);
    float* Kb  = (float*)alloc(NC * 4);
    float* Vb  = (float*)alloc(NC * 4);

    float* HW1 = (float*)alloc((size_t)ROWS * DW_ * 4);
    float* HA1 = (float*)alloc((size_t)ROWS * DA_ * 4);
    float* HV1 = (float*)alloc((size_t)ROWS * DV_ * 4);
    float* HG1 = (float*)alloc((size_t)ROWS * DG_ * 4);

    unsigned short* HWb = (unsigned short*)alloc((size_t)ROWS * DW_ * 2);
    unsigned short* HAb = (unsigned short*)alloc((size_t)ROWS * DA_ * 2);
    unsigned short* HVb = (unsigned short*)alloc((size_t)ROWS * DV_ * 2);
    unsigned short* HGb = (unsigned short*)alloc((size_t)ROWS * DG_ * 2);

    float* W2b = (float*)alloc(NC * 4);   // hw2 -> decay (in place)
    float* A2b = (float*)alloc(NC * 4);   // ha2 -> a (in place)
    float* V2b = (float*)alloc(NC * 4);   // hv2
    float* Gb  = (float*)alloc(NC * 4);   // gate
    float* KKb = (float*)alloc(NC * 4);
    float* Yb  = (float*)alloc(NC * 4);
    unsigned short* OUTG = (unsigned short*)alloc(NC * 2);

    // 1) token-shift mixing -> bf16 activations
    mix_kernel<<<(unsigned)(NC / 256), 256, 0, stream>>>(
        x, x_r, x_w, x_k, x_v, x_a, x_g, XR, XW, XK, XV, XA, XG);

    // 2) weight conversion to bf16 ([N,K] row-major for the NT GEMM)
    launch_cvt(Wr, WrB, Cc, Cc, 0, stream);
    launch_cvt(Wk, WkB, Cc, Cc, 0, stream);
    launch_cvt(Wv, WvB, Cc, Cc, 0, stream);
    launch_cvt(Wo, WoB, Cc, Cc, 0, stream);
    launch_cvt(w1, w1B, Cc, DW_, 1, stream);   // (C,DW) -> [DW,C]
    launch_cvt(w2, w2B, DW_, Cc, 1, stream);   // (DW,C) -> [C,DW]
    launch_cvt(a1, a1B, Cc, DA_, 1, stream);
    launch_cvt(a2, a2B, DA_, Cc, 1, stream);
    launch_cvt(v1, v1B, Cc, DV_, 1, stream);
    launch_cvt(v2, v2B, DV_, Cc, 1, stream);
    launch_cvt(g1, g1B, Cc, DG_, 1, stream);
    launch_cvt(g2, g2B, DG_, Cc, 1, stream);

    // 3) big projections: r = xr@Wr^T, k = xk@Wk^T, v = xv@Wv^T
    launch_gemm(XR, WrB, Rb, ROWS, Cc, Cc, stream);
    launch_gemm(XK, WkB, Kb, ROWS, Cc, Cc, stream);
    launch_gemm(XV, WvB, Vb, ROWS, Cc, Cc, stream);

    // 4) LoRA "up" GEMMs
    launch_gemm(XW, w1B, HW1, ROWS, DW_, Cc, stream);
    launch_gemm(XA, a1B, HA1, ROWS, DA_, Cc, stream);
    launch_gemm(XV, v1B, HV1, ROWS, DV_, Cc, stream);
    launch_gemm(XG, g1B, HG1, ROWS, DG_, Cc, stream);

    // 5) inner activations -> bf16 (w: tanh, a: none, v: none, g: sigmoid)
    act_bf16_kernel<<<(unsigned)(((size_t)ROWS * DW_ + 255) / 256), 256, 0, stream>>>(HW1, HWb, (size_t)ROWS * DW_, 1);
    act_bf16_kernel<<<(unsigned)(((size_t)ROWS * DA_ + 255) / 256), 256, 0, stream>>>(HA1, HAb, (size_t)ROWS * DA_, 0);
    act_bf16_kernel<<<(unsigned)(((size_t)ROWS * DV_ + 255) / 256), 256, 0, stream>>>(HV1, HVb, (size_t)ROWS * DV_, 0);
    act_bf16_kernel<<<(unsigned)(((size_t)ROWS * DG_ + 255) / 256), 256, 0, stream>>>(HG1, HGb, (size_t)ROWS * DG_, 2);

    // 6) LoRA "down" GEMMs
    launch_gemm(HWb, w2B, W2b, ROWS, Cc, DW_, stream);
    launch_gemm(HAb, a2B, A2b, ROWS, Cc, DA_, stream);
    launch_gemm(HVb, v2B, V2b, ROWS, Cc, DV_, stream);
    launch_gemm(HGb, g2B, Gb,  ROWS, Cc, DG_, stream);

    // 7) fused pointwise stage (decay / a / v-residual / kk-norm / k-scale)
    post_kernel<<<dim3(Hh, ROWS), 64, 0, stream>>>(
        W2b, A2b, V2b, Kb, Vb, vfirst, KKb, w0, a0, v0, k_k, k_a);

    // 8) WKV7 sequential scan over T (128 independent head-scans)
    wkv_kernel<<<dim3(Hh, Bc), 64, 0, stream>>>(Rb, W2b, Kb, Vb, KKb, A2b, Yb);

    // 9) groupnorm + bonus + gate -> bf16
    gn_out_kernel<<<dim3(Hh, ROWS), 64, 0, stream>>>(
        Yb, Rb, Kb, Vb, Gb, r_k, ln_w, ln_b, OUTG);

    // 10) output projection straight into d_out (f32)
    launch_gemm(OUTG, WoB, out, ROWS, Cc, Cc, stream);
}
